// Encoder_28260884807881
// MI455X (gfx1250) — compile-verified, matrix operand order checked by reading, not run on hardware
//
#include <hip/hip_runtime.h>
#include <hip/hip_bf16.h>
#include <cstdint>
#include <cstddef>

// ---------------------------------------------------------------------------
// Stacked GRU (4 layers, reset_after=True), B=64 T=1024 D=U=512.
//  * whole input pre-converted to bf16 in parallel (off the critical path)
//  * per (t,layer): one fused kernel, 128 WGs x 8 waves,
//    v_wmma_f32_16x16x32_bf16 for both GEMMs, K split 8-way (12 WMMA/wave).
//    All 16 fragments staged in registers BEFORE the WMMA burst; a
//    sched_barrier(0) pins the phase boundary so the backend cannot sink the
//    loads into the WMMA region (which serialized every WMMA on loadcnt 0).
//  * LDS cross-wave reduce, gate math spread across all 8 waves.
//  * ping-pong fp32+bf16 hidden state in workspace.
// ---------------------------------------------------------------------------

typedef __bf16 bf16;
typedef __attribute__((ext_vector_type(16))) __bf16 v16bf;
typedef __attribute__((ext_vector_type(8)))  __bf16 v8bf;
typedef __attribute__((ext_vector_type(8)))  float  v8f;
typedef __attribute__((ext_vector_type(4)))  float  v4f;

#define B_  64
#define T_  1024
#define D_  512
#define U_  512
#define U3_ 1536
#define L_  4

__device__ __forceinline__ void sched_fence() {
#if __has_builtin(__builtin_amdgcn_sched_barrier)
    __builtin_amdgcn_sched_barrier(0);   // nothing may be moved across
#else
    asm volatile("" ::: "memory");
#endif
}

__device__ __forceinline__ bf16 f2bf(float f) {
    // round-to-nearest-even fp32 -> bf16
    uint32_t u = __builtin_bit_cast(uint32_t, f);
    u += 0x7FFFu + ((u >> 16) & 1u);
    uint16_t h = (uint16_t)(u >> 16);
    return __builtin_bit_cast(bf16, h);
}

__device__ __forceinline__ v16bf load_a16(const bf16* p) {
    // elements 0..7 = K kb..kb+7, 8..15 = K kb+16..kb+23 (ISA 16-bit A layout)
    v8bf lo = *reinterpret_cast<const v8bf*>(p);
    v8bf hi = *reinterpret_cast<const v8bf*>(p + 16);
    return __builtin_shufflevector(lo, hi, 0,1,2,3,4,5,6,7,8,9,10,11,12,13,14,15);
}

// ---------------------------------------------------------------------------
// Input conversion: x fp32 [B,T,D] -> bf16 [B,T,D], vectorized 4-wide.
// ---------------------------------------------------------------------------
__global__ __launch_bounds__(256) void cvt_x(
    const float* __restrict__ x, bf16* __restrict__ xb)
{
    const size_t total4 = (size_t)B_ * T_ * D_ / 4;
    size_t idx = (size_t)blockIdx.x * blockDim.x + threadIdx.x;
    if (idx >= total4) return;
    v4f v = *reinterpret_cast<const v4f*>(x + idx * 4);
    bf16* o = xb + idx * 4;
    o[0] = f2bf(v[0]); o[1] = f2bf(v[1]); o[2] = f2bf(v[2]); o[3] = f2bf(v[3]);
}

// ---------------------------------------------------------------------------
// Weight conversion: fp32 [L, K, 3U] -> bf16 transposed [L, 3U, K] (K-contig
// per output column -> B tiles load with the same K-striping as A tiles).
// D_ == U_ so one kernel handles both tensors.
// ---------------------------------------------------------------------------
__global__ __launch_bounds__(256) void cvt_weights(
    const float* __restrict__ Kc, const float* __restrict__ Rc,
    bf16* __restrict__ Wt, bf16* __restrict__ Rt)
{
    int idx = blockIdx.x * blockDim.x + threadIdx.x;
    const int total = L_ * D_ * U3_;
    if (idx >= total) return;
    int l   = idx / (D_ * U3_);
    int rem = idx % (D_ * U3_);
    int k   = rem / U3_;
    int n   = rem % U3_;
    Wt[(size_t)l * U3_ * D_ + (size_t)n * D_ + k] = f2bf(Kc[idx]);
    Rt[(size_t)l * U3_ * U_ + (size_t)n * U_ + k] = f2bf(Rc[idx]);
}

// ---------------------------------------------------------------------------
// One (timestep, layer) GRU step.
// Grid: 128 WGs (mt in 0..3 row-tiles of 16, nt in 0..31 unit-tiles).
// Block: 256 threads = 8 waves; wave w owns K chunk [w*64, w*64+64).
// Six f32 accumulators per WG: {z,r,h} x {input GEMM, recurrent GEMM}.
// ---------------------------------------------------------------------------
__global__ __launch_bounds__(256) void gru_step(
    const bf16*  __restrict__ in_act,  // layer input, bf16 (x_bf or prev h)
    int in_stride,                     // row stride: T*D for layer 0, else U
    const bf16*  __restrict__ h_rec,   // [B,U] bf16, this layer's old h
    const float* __restrict__ h_prev,  // [B,U] f32,  this layer's old h
    const bf16*  __restrict__ Wt,      // [3U,D] bf16 transposed input weights
    const bf16*  __restrict__ Rt,      // [3U,U] bf16 transposed recurrent wts
    const float* __restrict__ bias,    // [2,3U] fp32
    bf16*  __restrict__ h_out_bf,      // [B,U]
    float* __restrict__ h_out_f,       // [B,U]
    float* __restrict__ out_t)         // &out[t*U], batch stride T*U, or null
{
    const int wave = threadIdx.x >> 5;    // 0..7
    const int lane = threadIdx.x & 31;
    const int mt   = blockIdx.x >> 5;     // 0..3
    const int nt   = blockIdx.x & 31;     // 0..31

    const int m    = mt * 16 + (lane & 15);   // batch row for A tiles
    const int n    = nt * 16 + (lane & 15);   // unit column for B tiles
    const int kgrp = (lane >> 4) * 8;         // ISA A-layout lane half

    __shared__ float red[8][6][256];          // 48 KB cross-wave reduction

    // ---- phase 1: issue ALL fragment loads (kept live in distinct regs) --
    v16bf a_in[2], a_h[2], bW[2][3], bR[2][3];
    #pragma unroll
    for (int i = 0; i < 2; ++i) {
        const int kk = wave * 2 + i;          // 0..15  (K step of 32)
        const int kb = kk * 32 + kgrp;
        a_in[i] = load_a16(in_act + (size_t)m * in_stride + kb);
        a_h[i]  = load_a16(h_rec  + (size_t)m * U_        + kb);
        #pragma unroll
        for (int g = 0; g < 3; ++g) {
            bW[i][g] = load_a16(Wt + (size_t)(g * U_ + n) * D_ + kb);
            bR[i][g] = load_a16(Rt + (size_t)(g * U_ + n) * U_ + kb);
        }
    }

    sched_fence();   // loads may not be sunk past this point

    // ---- phase 2: WMMA burst (waits stagger as loads return in order) ----
    v8f accW[3] = {v8f{}, v8f{}, v8f{}};
    v8f accR[3] = {v8f{}, v8f{}, v8f{}};
    #pragma unroll
    for (int i = 0; i < 2; ++i) {
        #pragma unroll
        for (int g = 0; g < 3; ++g) {
            accW[g] = __builtin_amdgcn_wmma_f32_16x16x32_bf16(
                false, a_in[i], false, bW[i][g], (short)0, accW[g], false, false);
            accR[g] = __builtin_amdgcn_wmma_f32_16x16x32_bf16(
                false, a_h[i],  false, bR[i][g], (short)0, accR[g], false, false);
        }
    }

    sched_fence();   // keep LDS-store tail out of the load window

    #pragma unroll
    for (int g = 0; g < 3; ++g) {
        #pragma unroll
        for (int r = 0; r < 8; ++r) {
            red[wave][g][r * 32 + lane]     = accW[g][r];
            red[wave][3 + g][r * 32 + lane] = accR[g][r];
        }
    }
    __syncthreads();

    // Gate math: wave w handles accumulator row r = w (all 8 waves active).
    {
        const int r = wave;
        const int u = n;
        const int o = r * 32 + lane;
        const float b0z = bias[u],          b1z = bias[U3_ + u];
        const float b0r = bias[U_ + u],     b1r = bias[U3_ + U_ + u];
        const float b0h = bias[2 * U_ + u], b1h = bias[U3_ + 2 * U_ + u];

        float xz = 0.f, xr = 0.f, xh = 0.f, rz = 0.f, rr = 0.f, rh = 0.f;
        #pragma unroll
        for (int w = 0; w < 8; ++w) {
            xz += red[w][0][o]; xr += red[w][1][o]; xh += red[w][2][o];
            rz += red[w][3][o]; rr += red[w][4][o]; rh += red[w][5][o];
        }

        const int row = mt * 16 + ((lane < 16) ? r : r + 8);
        const float z  = 1.f / (1.f + __expf(-(xz + rz + b0z + b1z)));
        const float rg = 1.f / (1.f + __expf(-(xr + rr + b0r + b1r)));
        const float hh = tanhf(xh + b0h + rg * (rh + b1h));
        const float hp = h_prev[(size_t)row * U_ + u];
        const float hn = z * hp + (1.f - z) * hh;

        h_out_f [(size_t)row * U_ + u] = hn;
        h_out_bf[(size_t)row * U_ + u] = f2bf(hn);
        if (out_t) out_t[(size_t)row * (T_ * U_) + u] = hn;
    }
}

// ---------------------------------------------------------------------------
extern "C" void kernel_launch(void* const* d_in, const int* in_sizes, int n_in,
                              void* d_out, int out_size, void* d_ws, size_t ws_size,
                              hipStream_t stream) {
    (void)in_sizes; (void)n_in; (void)out_size; (void)ws_size;

    const float* x  = (const float*)d_in[0];   // [B,T,D]
    const float* Kc = (const float*)d_in[1];   // [L,D,3U]
    const float* Rc = (const float*)d_in[2];   // [L,U,3U]
    const float* Bc = (const float*)d_in[3];   // [L,2,3U]
    float* out = (float*)d_out;                // [B,T,U] ++ [L,B,U]

    char* ws = (char*)d_ws;
    const size_t wt_bytes = (size_t)L_ * U3_ * D_ * sizeof(bf16);   // 6.3 MB
    const size_t st_elems = (size_t)L_ * B_ * U_;                   // per buffer
    const size_t x_elems  = (size_t)B_ * T_ * D_;
    bf16*  Wt  = (bf16*)ws;  ws += wt_bytes;
    bf16*  Rt  = (bf16*)ws;  ws += wt_bytes;
    bf16*  xb  = (bf16*)ws;  ws += x_elems * sizeof(bf16);          // 64 MB
    bf16*  hbf = (bf16*)ws;  ws += 2 * st_elems * sizeof(bf16);     // ping-pong
    float* hfp = (float*)ws; ws += 2 * st_elems * sizeof(float);    // ping-pong

    // zero initial states (deterministic per call)
    hipMemsetAsync(hbf, 0, 2 * st_elems * sizeof(bf16),  stream);
    hipMemsetAsync(hfp, 0, 2 * st_elems * sizeof(float), stream);

    // parallel precompute: bf16 input + transposed bf16 weights
    {
        const size_t n4 = x_elems / 4;
        cvt_x<<<(unsigned)((n4 + 255) / 256), 256, 0, stream>>>(x, xb);
        const int total = L_ * D_ * U3_;
        cvt_weights<<<(total + 255) / 256, 256, 0, stream>>>(Kc, Rc, Wt, Rt);
    }

    const size_t hs = (size_t)B_ * U_;   // per-layer state elements
    for (int t = 0; t < T_; ++t) {
        const int cur = t & 1, nxt = cur ^ 1;
        for (int l = 0; l < L_; ++l) {
            const bf16* in_act = (l == 0)
                ? xb + (size_t)t * D_
                : hbf + ((size_t)nxt * L_ + (l - 1)) * hs;
            const int in_stride = (l == 0) ? (T_ * D_) : U_;
            gru_step<<<128, 256, 0, stream>>>(
                in_act, in_stride,
                hbf + ((size_t)cur * L_ + l) * hs,
                hfp + ((size_t)cur * L_ + l) * hs,
                Wt + (size_t)l * U3_ * D_,
                Rt + (size_t)l * U3_ * U_,
                Bc + (size_t)l * 2 * U3_,
                hbf + ((size_t)nxt * L_ + l) * hs,
                hfp + ((size_t)nxt * L_ + l) * hs,
                (l == L_ - 1) ? out + (size_t)t * U_ : nullptr);
        }
    }

    // T=1024: last step wrote buffer 0 -> final states [L,B,U] are contiguous
    hipMemcpyAsync(out + (size_t)B_ * T_ * U_, hfp,
                   st_elems * sizeof(float), hipMemcpyDeviceToDevice, stream);
}
